// Set2Set_48902497632442
// MI455X (gfx1250) — matrix operand order repaired and checked by reading, not compiled, not run on hardware
//
#include <hip/hip_runtime.h>
#include <hip/hip_bf16.h>
#include <cstdint>

// Problem constants (match reference)
#define DIMD   256          // D
#define K_IH   512          // 2D
#define NG4    1024         // 4D
// B = 16384 graphs, N = 400000 nodes (taken from in_sizes/out_size)

typedef __attribute__((ext_vector_type(8)))  float         v8f;
typedef __attribute__((ext_vector_type(16))) __bf16        v16bf;
typedef __attribute__((ext_vector_type(8)))  __bf16        v8bf;
typedef __attribute__((ext_vector_type(4)))  unsigned int  v4u;

#define CAT16(lo,hi) __builtin_shufflevector(lo,hi,0,1,2,3,4,5,6,7,8,9,10,11,12,13,14,15)

static __device__ __forceinline__ float sigmoidf_(float x) {
    return 1.0f / (1.0f + __expf(-x));
}

// ---------------------------------------------------------------------------
// Prep: cast weights to bf16 (kept in native [4D,K] row-major so B-operand
// lanes read contiguous K), fuse the two bias vectors.
// ---------------------------------------------------------------------------
__global__ void prep_kernel(const float* __restrict__ Wih, const float* __restrict__ Whh,
                            const float* __restrict__ bih, const float* __restrict__ bhh,
                            __bf16* __restrict__ wihb, __bf16* __restrict__ whhb,
                            float* __restrict__ bias)
{
    int i = blockIdx.x * blockDim.x + threadIdx.x;
    int stride = gridDim.x * blockDim.x;
    for (int t = i; t < NG4 * K_IH; t += stride) wihb[t] = (__bf16)Wih[t];
    for (int t = i; t < NG4 * DIMD; t += stride) whhb[t] = (__bf16)Whh[t];
    for (int t = i; t < NG4;        t += stride) bias[t] = bih[t] + bhh[t];
}

// Zero a 16B-granular region (c, h_bf16 double buffer, q_star_bf16).
__global__ void zero_kernel(v4u* __restrict__ p, long long n16)
{
    long long i = (long long)blockIdx.x * blockDim.x + threadIdx.x;
    long long stride = (long long)gridDim.x * blockDim.x;
    v4u z = {0u, 0u, 0u, 0u};
    for (; i < n16; i += stride) p[i] = z;
}

// Segment boundaries from the sorted batch vector: segp[b]..segp[b+1] is graph b.
__global__ void segptr_kernel(const int* __restrict__ batch, int* __restrict__ segp,
                              int N, int B)
{
    int i = blockIdx.x * blockDim.x + threadIdx.x;
    if (i >= N) return;
    int bi = batch[i];
    int bp = (i == 0) ? -1 : batch[i - 1];
    for (int b = bp + 1; b <= bi; ++b) segp[b] = i;
    if (i == N - 1) for (int b = bi + 1; b <= B; ++b) segp[b] = N;
}

// ---------------------------------------------------------------------------
// Fused LSTM step: gates = qsb(bf16)@Wih^T + hbp(bf16)@Whh^T + bias, then the
// full cell update in registers. One wave per 16x16 (b,d) tile, 4 gate
// accumulators. 24 K-iterations x 4 WMMA = 96 v_wmma per wave.
// ---------------------------------------------------------------------------
__global__ __launch_bounds__(256) void lstm_kernel(
    const __bf16* __restrict__ qsb,   // [B, 512] q_star bf16 (t-1)
    const __bf16* __restrict__ hbp,   // [B, 256] h bf16      (t-1)
    const __bf16* __restrict__ wih,   // [1024, 512]
    const __bf16* __restrict__ whh,   // [1024, 256]
    const float*  __restrict__ bias,  // [1024]
    float* __restrict__ h,            // [B, 256] out (f32, for attention)
    float* __restrict__ c,            // [B, 256] in/out
    __bf16* __restrict__ hbn)         // [B, 256] out (bf16, next step GEMM)
{
    const int lane = threadIdx.x & 31;
    const int wv   = threadIdx.x >> 5;
    const int gw   = blockIdx.x * 8 + wv;
    const int m0   = (gw >> 4) * 16;   // batch-row tile
    const int d0   = (gw & 15) * 16;   // hidden-dim tile (within D=256)
    const int ln   = lane & 15;
    const int half = lane >> 4;

    v8f acc0 = {}, acc1 = {}, acc2 = {}, acc3 = {};

    // ---- q_star @ W_ih^T  (K = 512) ----
    {
        const __bf16* arow = qsb + (size_t)(m0 + ln) * K_IH;
        const __bf16* w0 = wih + (size_t)(0 * DIMD + d0 + ln) * K_IH;
        const __bf16* w1 = wih + (size_t)(1 * DIMD + d0 + ln) * K_IH;
        const __bf16* w2 = wih + (size_t)(2 * DIMD + d0 + ln) * K_IH;
        const __bf16* w3 = wih + (size_t)(3 * DIMD + d0 + ln) * K_IH;
#pragma unroll 4
        for (int k0 = 0; k0 < K_IH; k0 += 32) {
            v8bf alo = *(const v8bf*)(arow + k0 + half * 16);
            v8bf ahi = *(const v8bf*)(arow + k0 + half * 16 + 8);
            v16bf a  = CAT16(alo, ahi);
            v16bf b0 = *(const v16bf*)(w0 + k0 + half * 16);
            v16bf b1 = *(const v16bf*)(w1 + k0 + half * 16);
            v16bf b2 = *(const v16bf*)(w2 + k0 + half * 16);
            v16bf b3 = *(const v16bf*)(w3 + k0 + half * 16);
            acc0 = __builtin_amdgcn_wmma_f32_16x16x32_bf16(false, a, false, b0, (short)0, acc0, false, false);
            acc1 = __builtin_amdgcn_wmma_f32_16x16x32_bf16(false, a, false, b1, (short)0, acc1, false, false);
            acc2 = __builtin_amdgcn_wmma_f32_16x16x32_bf16(false, a, false, b2, (short)0, acc2, false, false);
            acc3 = __builtin_amdgcn_wmma_f32_16x16x32_bf16(false, a, false, b3, (short)0, acc3, false, false);
        }
    }
    // ---- h_{t-1} @ W_hh^T  (K = 256) ----
    {
        const __bf16* arow = hbp + (size_t)(m0 + ln) * DIMD;
        const __bf16* w0 = whh + (size_t)(0 * DIMD + d0 + ln) * DIMD;
        const __bf16* w1 = whh + (size_t)(1 * DIMD + d0 + ln) * DIMD;
        const __bf16* w2 = whh + (size_t)(2 * DIMD + d0 + ln) * DIMD;
        const __bf16* w3 = whh + (size_t)(3 * DIMD + d0 + ln) * DIMD;
#pragma unroll 4
        for (int k0 = 0; k0 < DIMD; k0 += 32) {
            v8bf alo = *(const v8bf*)(arow + k0 + half * 16);
            v8bf ahi = *(const v8bf*)(arow + k0 + half * 16 + 8);
            v16bf a  = CAT16(alo, ahi);
            v16bf b0 = *(const v16bf*)(w0 + k0 + half * 16);
            v16bf b1 = *(const v16bf*)(w1 + k0 + half * 16);
            v16bf b2 = *(const v16bf*)(w2 + k0 + half * 16);
            v16bf b3 = *(const v16bf*)(w3 + k0 + half * 16);
            acc0 = __builtin_amdgcn_wmma_f32_16x16x32_bf16(false, a, false, b0, (short)0, acc0, false, false);
            acc1 = __builtin_amdgcn_wmma_f32_16x16x32_bf16(false, a, false, b1, (short)0, acc1, false, false);
            acc2 = __builtin_amdgcn_wmma_f32_16x16x32_bf16(false, a, false, b2, (short)0, acc2, false, false);
            acc3 = __builtin_amdgcn_wmma_f32_16x16x32_bf16(false, a, false, b3, (short)0, acc3, false, false);
        }
    }

    // ---- LSTM cell update in registers (gates i,f,g,o = acc0..3) ----
    const int dcol = d0 + ln;
    const float bi0 = bias[0 * DIMD + dcol];
    const float bi1 = bias[1 * DIMD + dcol];
    const float bi2 = bias[2 * DIMD + dcol];
    const float bi3 = bias[3 * DIMD + dcol];
#pragma unroll
    for (int rr = 0; rr < 8; ++rr) {
        const int brow = m0 + rr + 8 * half;          // C-layout: M = r + 8*(lane/16)
        const size_t idx = (size_t)brow * DIMD + dcol;
        float gi = sigmoidf_(acc0[rr] + bi0);
        float gf = sigmoidf_(acc1[rr] + bi1);
        float gg = tanhf(acc2[rr] + bi2);
        float go = sigmoidf_(acc3[rr] + bi3);
        float cn = gf * c[idx] + gi * gg;
        float hn = go * tanhf(cn);
        c[idx]   = cn;
        h[idx]   = hn;
        hbn[idx] = (__bf16)hn;
    }
}

// ---------------------------------------------------------------------------
// Fused attention: one wave per graph, single pass over x with online softmax.
// lane l owns channels [8l, 8l+8). Writes q_star (f32 -> d_out, bf16 -> qsb).
// ---------------------------------------------------------------------------
__global__ __launch_bounds__(256) void attn_kernel(
    const float* __restrict__ x,      // [N, 256]
    const int*   __restrict__ segp,   // [B+1]
    const float* __restrict__ h,      // [B, 256]
    float*  __restrict__ qstar,       // [B, 512] == d_out
    __bf16* __restrict__ qsb)         // [B, 512]
{
    const int lane = threadIdx.x & 31;
    const int b    = blockIdx.x * 8 + (threadIdx.x >> 5);
    const int s = segp[b];
    const int e = segp[b + 1];

    v8f q = *(const v8f*)(h + (size_t)b * DIMD + lane * 8);
    v8f r = {};
    float m = -__builtin_inff();
    float z = 0.0f;

    for (int i = s; i < e; ++i) {
        if (i + 4 < e)
            __builtin_prefetch(x + (size_t)(i + 4) * DIMD + lane * 8, 0, 1);
        v8f xv = *(const v8f*)(x + (size_t)i * DIMD + lane * 8);
        float d = 0.0f;
#pragma unroll
        for (int t = 0; t < 8; ++t) d += q[t] * xv[t];
#pragma unroll
        for (int off = 1; off < 32; off <<= 1) d += __shfl_xor(d, off, 32);
        float mn = fmaxf(m, d);
        float sc = __expf(m - mn);    // 0 on first iteration (m = -inf)
        float w  = __expf(d - mn);
        z = z * sc + w;
#pragma unroll
        for (int t = 0; t < 8; ++t) r[t] = r[t] * sc + w * xv[t];
        m = mn;
    }

    const float inv = 1.0f / (z + 1e-10f);
    v8f ro;
    v8bf qb, rb;
#pragma unroll
    for (int t = 0; t < 8; ++t) {
        ro[t] = r[t] * inv;
        qb[t] = (__bf16)q[t];
        rb[t] = (__bf16)ro[t];
    }
    *(v8f*)(qstar + (size_t)b * 512 + lane * 8)        = q;
    *(v8f*)(qstar + (size_t)b * 512 + 256 + lane * 8)  = ro;
    *(v8bf*)(qsb  + (size_t)b * 512 + lane * 8)        = qb;
    *(v8bf*)(qsb  + (size_t)b * 512 + 256 + lane * 8)  = rb;
}

// ---------------------------------------------------------------------------
extern "C" void kernel_launch(void* const* d_in, const int* in_sizes, int n_in,
                              void* d_out, int out_size, void* d_ws, size_t ws_size,
                              hipStream_t stream)
{
    const float* x     = (const float*)d_in[0];
    const int*   batch = (const int*)d_in[1];
    const float* Wih   = (const float*)d_in[2];
    const float* Whh   = (const float*)d_in[3];
    const float* bih   = (const float*)d_in[4];
    const float* bhh   = (const float*)d_in[5];
    float* qstar = (float*)d_out;

    const int N = in_sizes[1];        // 400000
    const int B = out_size / 512;     // 16384

    // Workspace carving (256B aligned slabs)
    auto au = [](size_t v) { return (v + 255) & ~(size_t)255; };
    char* p = (char*)d_ws;
    size_t off = 0;
    __bf16* wihb = (__bf16*)(p + off); off = au(off + (size_t)NG4 * K_IH * 2);
    __bf16* whhb = (__bf16*)(p + off); off = au(off + (size_t)NG4 * DIMD * 2);
    float*  bias = (float*)(p + off);  off = au(off + (size_t)NG4 * 4);
    int*    segp = (int*)(p + off);    off = au(off + (size_t)(B + 1) * 4);
    float*  h    = (float*)(p + off);  off = au(off + (size_t)B * DIMD * 4);
    char*   zb   = p + off;            // zero region starts here
    float*  c    = (float*)(p + off);  off = au(off + (size_t)B * DIMD * 4);
    __bf16* hb0  = (__bf16*)(p + off); off = au(off + (size_t)B * DIMD * 2);
    __bf16* hb1  = (__bf16*)(p + off); off = au(off + (size_t)B * DIMD * 2);
    __bf16* qsb  = (__bf16*)(p + off); off = au(off + (size_t)B * 512 * 2);
    const long long zbytes = (long long)((p + off) - zb);

    prep_kernel<<<2048, 256, 0, stream>>>(Wih, Whh, bih, bhh, wihb, whhb, bias);
    zero_kernel<<<2048, 256, 0, stream>>>((v4u*)zb, zbytes / 16);
    segptr_kernel<<<(N + 255) / 256, 256, 0, stream>>>(batch, segp, N, B);

    __bf16* hb[2] = {hb0, hb1};
    const int lstm_blocks = (B / 16) * 16 / 8;   // one wave per 16x16 tile, D/16=16 tiles
    const int attn_blocks = B / 8;               // one wave per graph
    for (int s = 0; s < 3; ++s) {
        lstm_kernel<<<lstm_blocks, 256, 0, stream>>>(qsb, hb[s & 1], wihb, whhb, bias,
                                                     h, c, hb[(s + 1) & 1]);
        attn_kernel<<<attn_blocks, 256, 0, stream>>>(x, segp, h, qstar, qsb);
    }
}
